// CRF_25031069401437
// MI455X (gfx1250) — matrix operand order repaired and checked by reading, not compiled, not run on hardware
//
#include <hip/hip_runtime.h>
#include <math.h>

typedef _Float16 v16h __attribute__((ext_vector_type(16)));
typedef _Float16 h8   __attribute__((ext_vector_type(8)));
typedef float    v8f  __attribute__((ext_vector_type(8)));

#define Bn 512
#define Ln 512
#define Nn 128
#define TB 16      // batch rows per block (WMMA M tile)
#define NT 256     // 8 wave32 per block; each wave owns one 16-wide N tile
#define PPAD 8     // pad P rows by 8 halfs (16B) -> breaks LDS bank conflicts
#define SPAD 4
#define EPAD 4     // pad emission-tile rows by 4 floats

#define LOG2E 1.4426950408889634f
#define LN2   0.6931471805599453f

// Raw hardware transcendentals (v_exp_f32 = 2^x, v_log_f32 = log2 x).
// Inputs here never hit the denorm-fixup cases (acc >= 0.9; exp underflow->0 is fine),
// so skip the precise-path cmp/cndmask/ldexp chains. Fallback: standard exp2f/log2f
// (valid in BOTH host and device passes; amdgcn builtins only exist on the device pass).
#if __has_builtin(__builtin_amdgcn_exp2f)
#define FAST_EXP2(x) __builtin_amdgcn_exp2f(x)
#else
#define FAST_EXP2(x) exp2f(x)
#endif
#if __has_builtin(__builtin_amdgcn_logf)
#define FAST_LOG2(x) __builtin_amdgcn_logf(x)
#else
#define FAST_LOG2(x) log2f(x)
#endif

// CDNA5 async global->LDS copy (ASYNCcnt-tracked), saddr-null ("off") form.
__device__ __forceinline__ void async_load_b128(unsigned lds_off, const void* gaddr) {
    asm volatile("global_load_async_to_lds_b128 %0, %1, off"
                 :: "v"(lds_off), "v"((unsigned long long)(uintptr_t)gaddr)
                 : "memory");
}

// ---------------------------------------------------------------------------
// Forward (normalizer) scan:  S' = mask*(log(exp(S-m) @ exp(T)) + m + em) + (1-mask)*S
// exp(T) lives in registers as WMMA B-fragments for all 511 steps; emission is
// double-buffered in LDS via global_load_async_to_lds_b128.
// ---------------------------------------------------------------------------
__global__ __launch_bounds__(NT)
void crf_forward_kernel(const float* __restrict__ emission,
                        const float* __restrict__ mask,
                        const float* __restrict__ start_trans,
                        const float* __restrict__ trans,
                        float* __restrict__ norm_out)
{
    __shared__ float     Ssc[TB][Nn + SPAD];   // running scores (f32)
    __shared__ _Float16  Pm [TB][Nn + PPAD];   // exp(S - m) in f16 (WMMA A operand)
    __shared__ float     Em [2][TB][Nn + EPAD];// double-buffered emission tile
    __shared__ float     red[TB][16];          // per-row reduction scratch
    __shared__ float     mrow[TB];
    __shared__ float     maskrow[TB];

    const int tid  = threadIdx.x;
    const int lane = tid & 31;
    const int w    = tid >> 5;        // wave id -> N tile [16w, 16w+16)
    const int r    = tid >> 4;        // elementwise phases: row 0..15
    const int s    = tid & 15;        //                     16 threads per row
    const int b0   = blockIdx.x * TB;

    // ---- B fragments: E[k][n] = exp(trans[k][n]) in f16, ISA B layout -------
    // lanes 0-15: K=0..7,16..23 of the 32-slice; lanes 16-31: K=8..15,24..31. n = 16w + lane%16.
    const int ncol  = (w << 4) + (lane & 15);
    const int khalf = (lane >> 4) << 3;        // 0 or 8
    v16h bfrag[4];
#pragma unroll
    for (int kk = 0; kk < 4; ++kk) {
#pragma unroll
        for (int e = 0; e < 16; ++e) {
            const int k = kk * 32 + khalf + (e & 7) + ((e >> 3) << 4);
            bfrag[kk][e] = (_Float16)FAST_EXP2(trans[k * Nn + ncol] * LOG2E);
        }
    }

    // ---- t = 0: S = start_trans + emission[:,0,:] ---------------------------
    for (int k = 0; k < 8; ++k) {
        const int c = s + (k << 4);
        Ssc[r][c] = start_trans[c] + emission[(size_t)(b0 + r) * Ln * Nn + c];
    }

    // ---- prime the async pipeline: emission tile for t = 1 -> Em[1] ---------
    {
        const int cc = s << 3;                 // 8 floats = two b128 per thread
        const float* g = &emission[(size_t)(b0 + r) * Ln * Nn + (size_t)1 * Nn + cc];
        const unsigned l = (unsigned)(uintptr_t)&Em[1][r][cc];
        async_load_b128(l,      g);
        async_load_b128(l + 16, (const char*)g + 16);
    }
    __syncthreads();

    // A fragment addressing (16-bit A 16x32 layout)
    const int mA    = lane & 15;               // row (batch within tile)
    const int koffA = (lane >> 4) << 3;        // lane half: +0 or +8 within 32-slice
    const int nlo   = (w << 4) + (lane & 15);  // D: column
    const int mhi   = (lane >> 4) << 3;        // D: row base (0 or 8)

    for (int t = 1; t < Ln; ++t) {
        // issue async loads for the NEXT timestep's emission tile
        if (t + 1 < Ln) {
            const int cc = s << 3;
            const float* g = &emission[(size_t)(b0 + r) * Ln * Nn + (size_t)(t + 1) * Nn + cc];
            const unsigned l = (unsigned)(uintptr_t)&Em[(t + 1) & 1][r][cc];
            async_load_b128(l,      g);
            async_load_b128(l + 16, (const char*)g + 16);
        }

        // phase 1: per-row max partials + stage this step's mask
        float pmx = -__builtin_inff();
        for (int k = 0; k < 8; ++k) pmx = fmaxf(pmx, Ssc[r][s + (k << 4)]);
        red[r][s] = pmx;
        if (tid < TB) maskrow[tid] = mask[(size_t)(b0 + tid) * Ln + t];
        __syncthreads();
        if (tid < TB) {
            float m = red[tid][0];
            for (int i = 1; i < 16; ++i) m = fmaxf(m, red[tid][i]);
            mrow[tid] = m;
        }
        __syncthreads();

        // phase 2: P = exp2((S - m) * log2e) -> f16   (raw v_exp_f32)
        {
            const float m = mrow[r];
            for (int k = 0; k < 8; ++k) {
                const int c = s + (k << 4);
                Pm[r][c] = (_Float16)FAST_EXP2((Ssc[r][c] - m) * LOG2E);
            }
        }
        // current tile's async loads (issued last iteration) must have landed;
        // loads are in-order per wave: the 2 just-issued may stay outstanding.
        if (t + 1 < Ln) asm volatile("s_wait_asynccnt 0x2" ::: "memory");
        else            asm volatile("s_wait_asynccnt 0x0" ::: "memory");
        __syncthreads();   // publish Pm and Em[t&1] to all waves

        // phase 3: Z = P @ E via 4x V_WMMA_F32_16X16X32_F16 (K = 128)
        v8f acc = {};
#pragma unroll
        for (int kk = 0; kk < 4; ++kk) {
            const int kb = kk * 32 + koffA;
            const h8 lo = *(const h8*)&Pm[mA][kb];        // K = kb .. kb+7
            const h8 hi = *(const h8*)&Pm[mA][kb + 16];   // K = kb+16 .. kb+23
            v16h a;
#pragma unroll
            for (int e = 0; e < 8; ++e) { a[e] = lo[e]; a[e + 8] = hi[e]; }
            acc = __builtin_amdgcn_wmma_f32_16x16x32_f16(false, a, false, bfrag[kk],
                                                         (short)0, acc, false, false);
        }

        // phase 4: S = mask*(ln2*log2(Z) + m + em) + (1-mask)*S  (em from LDS tile)
#pragma unroll
        for (int v = 0; v < 8; ++v) {
            const int   mr = v + mhi;
            const float em = Em[t & 1][mr][nlo];
            const float sn = LN2 * FAST_LOG2(acc[v]) + mrow[mr] + em;
            const float mk = maskrow[mr];
            const float so = Ssc[mr][nlo];
            Ssc[mr][nlo] = mk * sn + (1.0f - mk) * so;
        }
        __syncthreads();
    }

    // ---- final logsumexp over states per batch row --------------------------
    float pmx = -__builtin_inff();
    for (int k = 0; k < 8; ++k) pmx = fmaxf(pmx, Ssc[r][s + (k << 4)]);
    red[r][s] = pmx;
    __syncthreads();
    if (tid < TB) {
        float m = red[tid][0];
        for (int i = 1; i < 16; ++i) m = fmaxf(m, red[tid][i]);
        mrow[tid] = m;
    }
    __syncthreads();
    {
        const float m = mrow[r];
        float ps = 0.0f;
        for (int k = 0; k < 8; ++k) ps += FAST_EXP2((Ssc[r][s + (k << 4)] - m) * LOG2E);
        red[r][s] = ps;
    }
    __syncthreads();
    if (tid < TB) {
        float sum = 0.0f;
        for (int i = 0; i < 16; ++i) sum += red[tid][i];
        norm_out[b0 + tid] = mrow[tid] + LN2 * FAST_LOG2(sum);
    }
}

// ---------------------------------------------------------------------------
// Path score: s0 + sum_t mask * (trans[prev,cur] + em[t,cur]); one block per batch
// ---------------------------------------------------------------------------
__global__ __launch_bounds__(128)
void crf_path_kernel(const float* __restrict__ emission,
                     const int*   __restrict__ target,
                     const float* __restrict__ mask,
                     const float* __restrict__ start_trans,
                     const float* __restrict__ trans,
                     float* __restrict__ path_out)
{
    __shared__ float red[128];
    const int b = blockIdx.x, tid = threadIdx.x;
    float acc = 0.0f;
    for (int t = 1 + tid; t < Ln; t += 128) {
        const int prev = target[(size_t)b * Ln + t - 1];
        const int cur  = target[(size_t)b * Ln + t];
        const float inc = trans[prev * Nn + cur]
                        + emission[(size_t)b * Ln * Nn + (size_t)t * Nn + cur];
        acc += mask[(size_t)b * Ln + t] * inc;
    }
    red[tid] = acc;
    __syncthreads();
    for (int off = 64; off > 0; off >>= 1) {
        if (tid < off) red[tid] += red[tid + off];
        __syncthreads();
    }
    if (tid == 0) {
        const int t0 = target[(size_t)b * Ln];
        path_out[b] = red[0] + start_trans[t0] + emission[(size_t)b * Ln * Nn + t0];
    }
}

// ---------------------------------------------------------------------------
// mean(normalizer - path)
// ---------------------------------------------------------------------------
__global__ __launch_bounds__(256)
void crf_final_kernel(const float* __restrict__ norm,
                      const float* __restrict__ path,
                      float* __restrict__ out)
{
    __shared__ float red[256];
    const int tid = threadIdx.x;
    float a = 0.0f;
    for (int b = tid; b < Bn; b += 256) a += norm[b] - path[b];
    red[tid] = a;
    __syncthreads();
    for (int off = 128; off > 0; off >>= 1) {
        if (tid < off) red[tid] += red[tid + off];
        __syncthreads();
    }
    if (tid == 0) out[0] = red[0] / (float)Bn;
}

extern "C" void kernel_launch(void* const* d_in, const int* in_sizes, int n_in,
                              void* d_out, int out_size, void* d_ws, size_t ws_size,
                              hipStream_t stream)
{
    const float* emission    = (const float*)d_in[0];
    const int*   target      = (const int*)  d_in[1];
    const float* mask        = (const float*)d_in[2];
    const float* start_trans = (const float*)d_in[3];
    const float* trans       = (const float*)d_in[4];
    float* out  = (float*)d_out;
    float* norm = (float*)d_ws;       // 512 floats
    float* path = norm + Bn;          // 512 floats

    crf_forward_kernel<<<Bn / TB, NT, 0, stream>>>(emission, mask, start_trans, trans, norm);
    crf_path_kernel  <<<Bn, 128, 0, stream>>>(emission, target, mask, start_trans, trans, path);
    crf_final_kernel <<<1, 256, 0, stream>>>(norm, path, out);
}
